// OneTransAttention_38190849196403
// MI455X (gfx1250) — matrix operand order repaired and checked by reading, not compiled, not run on hardware
//
#include <hip/hip_runtime.h>
#include <hip/hip_bf16.h>

// ---------------------------------------------------------------------------
// OneTrans attention for MI455X (gfx1250, wave32, WMMA bf16 / f32-accum)
//   B=4, L=2048, D=1024, H=16, Hd=64, L_ns read from device scalar.
// Workspace layout (needs 80 MB):
//   [ 0MB) q   bf16 [B,H,L,Hd]
//   [16MB) k   bf16 [B,H,L,Hd]
//   [32MB) v^T bf16 [B,H,Hd,L]
//   [48MB) attn_out f32 [B,L,D]
// ---------------------------------------------------------------------------

#define D_DIM 1024
#define L_SEQ 2048
#define BATCH 4
#define H_NUM 16
#define HDIM  64

typedef __bf16 bf16_t;
typedef __attribute__((ext_vector_type(16))) __bf16 v16bf;
typedef __attribute__((ext_vector_type(8)))  float  v8f;

union FragT {
  uint4    q[2];
  unsigned u[8];
  v16bf    v;
};

__device__ __forceinline__ unsigned short f2bf(float f) {
  unsigned u = __float_as_uint(f);
  unsigned r = u + 0x7FFFu + ((u >> 16) & 1u);   // round-to-nearest-even
  return (unsigned short)(r >> 16);
}
__device__ __forceinline__ unsigned pk2bf(float lo, float hi) {
  return (unsigned)f2bf(lo) | ((unsigned)f2bf(hi) << 16);
}

// ---------------------------------------------------------------------------
// GEMM: C[M=8192, N=1024] = A[M,K=1024](f32) * W[N,K](f32, row-major = B^T)
// mode 0: store bf16 into [B,H,L,Hd] (q / k)
// mode 1: store bf16 into [B,H,Hd,L] (v transposed)
// mode 2: store f32 + bias into [M,N] (output projection)
// Block: 128 threads = 4 waves. WG tile 64x64, K-step 64 (8 WMMAs/wave/step
// between barrier pairs), fp32->bf16 packed during LDS staging.
// ---------------------------------------------------------------------------
__global__ __launch_bounds__(128) void gemm_wmma(
    const float* __restrict__ A, const float* __restrict__ W,
    const float* __restrict__ bias,
    unsigned short* __restrict__ dst_bf, float* __restrict__ dst_f, int mode)
{
  __shared__ unsigned As[64][36];   // 64 bf16 (32 uints) + 4 pad, 144B rows
  __shared__ unsigned Bs[64][36];

  const int tid    = threadIdx.x;
  const int lane   = tid & 31;
  const int wave   = tid >> 5;
  const int half16 = (lane >= 16) ? 1 : 0;
  const int lm     = lane & 15;
  const int n0 = blockIdx.x * 64;
  const int m0 = blockIdx.y * 64;

  v8f acc[4];
  #pragma unroll
  for (int t = 0; t < 4; ++t) acc[t] = 0.f;

  const int srow = tid >> 1;            // staging row 0..63
  const int scol = (tid & 1) * 32;      // staging float col 0 / 32

  for (int k0 = 0; k0 < D_DIM; k0 += 64) {
    // ---- stage A and B 64x64 tiles (fp32 -> bf16) into LDS -----------------
    {
      const float4* ap = (const float4*)(A + (size_t)(m0 + srow) * D_DIM + k0 + scol);
      const float4* bp = (const float4*)(W + (size_t)(n0 + srow) * D_DIM + k0 + scol);
      if (k0 + 64 < D_DIM) {            // prefetch next K tile into WGP cache
        __builtin_prefetch(A + (size_t)(m0 + srow) * D_DIM + k0 + 64 + scol, 0, 3);
        __builtin_prefetch(W + (size_t)(n0 + srow) * D_DIM + k0 + 64 + scol, 0, 3);
      }
      #pragma unroll
      for (int i = 0; i < 8; ++i) {
        float4 fa = ap[i], fb = bp[i];
        As[srow][(scol >> 1) + 2 * i]     = pk2bf(fa.x, fa.y);
        As[srow][(scol >> 1) + 2 * i + 1] = pk2bf(fa.z, fa.w);
        Bs[srow][(scol >> 1) + 2 * i]     = pk2bf(fb.x, fb.y);
        Bs[srow][(scol >> 1) + 2 * i + 1] = pk2bf(fb.z, fb.w);
      }
    }
    __syncthreads();

    #pragma unroll
    for (int kk = 0; kk < 2; ++kk) {    // two K=32 sub-steps per staged tile
      // A frag: lane<16 -> k {0..7,16..23}, lane>=16 -> k {8..15,24..31}
      FragT a;
      const int arow = wave * 16 + lm;
      const int ka   = kk * 16 + half16 * 4;     // uint offset
      a.q[0] = *(const uint4*)&As[arow][ka];
      a.q[1] = *(const uint4*)&As[arow][ka + 8];

      // B frags: lane<16 -> k {0..7,16..23}, lane>=16 -> k {8..15,24..31}
      #pragma unroll
      for (int t = 0; t < 4; ++t) {
        FragT bfr;
        const int brow = t * 16 + lm;
        const int kb   = kk * 16 + half16 * 4;   // uint offset
        bfr.q[0] = *(const uint4*)&Bs[brow][kb];
        bfr.q[1] = *(const uint4*)&Bs[brow][kb + 8];
        acc[t] = __builtin_amdgcn_wmma_f32_16x16x32_bf16(
            false, a.v, false, bfr.v, (short)0, acc[t], false, false);
      }
    }
    __syncthreads();
  }

  // ---- store: C layout lane = n col, reg r = row (r or r+8 per half) -------
  #pragma unroll
  for (int t = 0; t < 4; ++t) {
    #pragma unroll
    for (int r = 0; r < 8; ++r) {
      const int gm = m0 + wave * 16 + r + half16 * 8;
      const int gn = n0 + t * 16 + lm;
      const float v = acc[t][r] + bias[gn];
      if (mode == 2) {
        dst_f[(size_t)gm * D_DIM + gn] = v;
      } else {
        const int b  = gm >> 11;                  // / L_SEQ
        const int l  = gm & (L_SEQ - 1);
        const int h  = gn >> 6;                   // / HDIM
        const int hd = gn & (HDIM - 1);
        const size_t idx = (mode == 0)
            ? ((((size_t)b * H_NUM + h) * L_SEQ + l) * HDIM + hd)
            : ((((size_t)b * H_NUM + h) * HDIM + hd) * L_SEQ + l);
        dst_bf[idx] = f2bf(v);
      }
    }
  }
}

// ---------------------------------------------------------------------------
// Flash attention: one (b,h); each wave owns a 16-row q tile. Computes
// S^T = K*Q^T tiles so softmax is per-lane; P^T C-layout of a 32-key pair is
// exactly the A-matrix VGPR layout for the P*V WMMA (no transpose needed).
// ---------------------------------------------------------------------------
__global__ __launch_bounds__(128) void attn_wmma(
    const unsigned short* __restrict__ qb, const unsigned short* __restrict__ kb,
    const unsigned short* __restrict__ vt, float* __restrict__ outp,
    const int* __restrict__ pLns)
{
  const int tid    = threadIdx.x;
  const int lane   = tid & 31;
  const int wave   = tid >> 5;
  const int half16 = (lane >= 16) ? 1 : 0;
  const int lm     = lane & 15;
  const int Lns    = *pLns;

  const int qbase = blockIdx.x * 64 + wave * 16;
  const int h     = blockIdx.y;
  const int b     = blockIdx.z;
  const size_t bh = (size_t)b * H_NUM + h;

  const unsigned short* qp = qb + bh * (size_t)L_SEQ * HDIM;
  const unsigned short* kp = kb + bh * (size_t)L_SEQ * HDIM;
  const unsigned short* vp = vt + bh * (size_t)HDIM * L_SEQ;

  // Q as B operand (persists): lane n -> q row qbase+lm, d chunks {8h..+7, 16+8h..+7}
  FragT qf[2];
  #pragma unroll
  for (int kk = 0; kk < 2; ++kk) {
    const unsigned short* p = qp + (size_t)(qbase + lm) * HDIM + kk * 32 + half16 * 8;
    qf[kk].q[0] = *(const uint4*)p;
    qf[kk].q[1] = *(const uint4*)(p + 16);
  }

  v8f o[4];
  #pragma unroll
  for (int t = 0; t < 4; ++t) o[t] = 0.f;
  float m_run = -1e30f, l_run = 0.f;
  const float scale = 0.125f;               // 1/sqrt(64)
  const int qg = qbase + lm;                // this lane's q column in S^T

  const int ntiles = (qbase < Lns) ? (L_SEQ / 16) : (qbase / 16 + 1);
  const int npairs = (ntiles + 1) >> 1;

  for (int p = 0; p < npairs; ++p) {
    const int j0 = 2 * p, j1 = 2 * p + 1;

    // K tiles as A operand: lane row = k local, d chunks {koff..+7, koff+16..+23}
    FragT ka0[2], ka1[2];
    #pragma unroll
    for (int kk = 0; kk < 2; ++kk) {
      const unsigned short* p0 = kp + (size_t)(j0 * 16 + lm) * HDIM + kk * 32 + half16 * 8;
      ka0[kk].q[0] = *(const uint4*)p0;
      ka0[kk].q[1] = *(const uint4*)(p0 + 16);
      const unsigned short* p1 = kp + (size_t)(j1 * 16 + lm) * HDIM + kk * 32 + half16 * 8;
      ka1[kk].q[0] = *(const uint4*)p1;
      ka1[kk].q[1] = *(const uint4*)(p1 + 16);
    }
    if (p + 1 < npairs)
      __builtin_prefetch(kp + (size_t)((j0 + 2) * 16 + lm) * HDIM, 0, 3);

    v8f s0 = 0.f, s1 = 0.f;
    s0 = __builtin_amdgcn_wmma_f32_16x16x32_bf16(false, ka0[0].v, false, qf[0].v, (short)0, s0, false, false);
    s0 = __builtin_amdgcn_wmma_f32_16x16x32_bf16(false, ka0[1].v, false, qf[1].v, (short)0, s0, false, false);
    s1 = __builtin_amdgcn_wmma_f32_16x16x32_bf16(false, ka1[0].v, false, qf[0].v, (short)0, s1, false, false);
    s1 = __builtin_amdgcn_wmma_f32_16x16x32_bf16(false, ka1[1].v, false, qf[1].v, (short)0, s1, false, false);

    // ---- online softmax (per-lane over 16 k rows) --------------------------
    float sv[16];
    float mt = -3.0e38f;
    #pragma unroll
    for (int r = 0; r < 8; ++r) {
      const int kg0 = j0 * 16 + r + half16 * 8;
      const int kg1 = j1 * 16 + r + half16 * 8;
      float a0 = s0[r] * scale;
      float a1 = s1[r] * scale;
      if (qg >= Lns && kg0 > qg) a0 = -3.0e38f;
      if (qg >= Lns && kg1 > qg) a1 = -3.0e38f;
      sv[r] = a0; sv[8 + r] = a1;
      mt = fmaxf(mt, fmaxf(a0, a1));
    }
    float mnew = fmaxf(m_run, mt);
    mnew = fmaxf(mnew, __shfl_xor(mnew, 16, 32));   // sync both lane halves
    const float alpha = __expf(m_run - mnew);

    float pe[16];
    float ls = 0.f;
    #pragma unroll
    for (int i = 0; i < 16; ++i) { pe[i] = __expf(sv[i] - mnew); ls += pe[i]; }
    l_run = l_run * alpha + ls;
    m_run = mnew;

    // P^T -> A frag, purely in-lane (matches 16-bit A layout for K=32 pair)
    FragT pf;
    #pragma unroll
    for (int j = 0; j < 8; ++j) pf.u[j] = pk2bf(pe[2 * j], pe[2 * j + 1]);

    // rescale O accumulators by alpha(q) (broadcast per C-layout row)
    #pragma unroll
    for (int r = 0; r < 8; ++r) {
      const float ar = __shfl(alpha, r + half16 * 8, 32);
      o[0][r] *= ar; o[1][r] *= ar; o[2][r] *= ar; o[3][r] *= ar;
    }

    // P*V: V^T rows are d, k contiguous -> direct B-operand loads
    #pragma unroll
    for (int t = 0; t < 4; ++t) {
      FragT vf;
      const unsigned short* vv = vp + (size_t)(t * 16 + lm) * L_SEQ + j0 * 16 + half16 * 8;
      vf.q[0] = *(const uint4*)vv;
      vf.q[1] = *(const uint4*)(vv + 16);
      o[t] = __builtin_amdgcn_wmma_f32_16x16x32_bf16(
          false, pf.v, false, vf.v, (short)0, o[t], false, false);
    }
  }

  // ---- finalize: combine lane halves' sums, normalize, store f32 ----------
  const float l_full = l_run + __shfl_xor(l_run, 16, 32);
  const float inv = 1.0f / l_full;
  #pragma unroll
  for (int r = 0; r < 8; ++r) {
    const float ir = __shfl(inv, r + half16 * 8, 32);
    const int grow = qbase + r + half16 * 8;
    float* dst = outp + ((size_t)b * L_SEQ + grow) * D_DIM + (size_t)h * HDIM;
    #pragma unroll
    for (int t = 0; t < 4; ++t) dst[t * 16 + lm] = o[t][r] * ir;
  }
}

// ---------------------------------------------------------------------------
extern "C" void kernel_launch(void* const* d_in, const int* in_sizes, int n_in,
                              void* d_out, int out_size, void* d_ws, size_t ws_size,
                              hipStream_t stream) {
  (void)in_sizes; (void)n_in; (void)out_size; (void)ws_size;
  const float* x   = (const float*)d_in[0];
  const int*   Lns = (const int*)  d_in[1];
  const float* Wq  = (const float*)d_in[2];
  const float* bq  = (const float*)d_in[3];
  const float* Wk  = (const float*)d_in[4];
  const float* bk  = (const float*)d_in[5];
  const float* Wv  = (const float*)d_in[6];
  const float* bv  = (const float*)d_in[7];
  const float* Wo  = (const float*)d_in[8];
  const float* bo  = (const float*)d_in[9];
  float* out = (float*)d_out;

  const size_t NE = (size_t)BATCH * L_SEQ * D_DIM;        // 8388608
  char* ws = (char*)d_ws;
  unsigned short* qb   = (unsigned short*)(ws);
  unsigned short* kb   = (unsigned short*)(ws + NE * 2);
  unsigned short* vt   = (unsigned short*)(ws + NE * 4);
  float*          attn = (float*)         (ws + NE * 6);

  dim3 gblk(128);
  dim3 ggrd(D_DIM / 64, (BATCH * L_SEQ) / 64);            // (16,128)

  gemm_wmma<<<ggrd, gblk, 0, stream>>>(x, Wq, bq, qb, nullptr, 0);
  gemm_wmma<<<ggrd, gblk, 0, stream>>>(x, Wk, bk, kb, nullptr, 0);
  gemm_wmma<<<ggrd, gblk, 0, stream>>>(x, Wv, bv, vt, nullptr, 1);

  attn_wmma<<<dim3(L_SEQ / 64, H_NUM, BATCH), gblk, 0, stream>>>(qb, kb, vt, attn, Lns);

  gemm_wmma<<<ggrd, gblk, 0, stream>>>(attn, Wo, bo, nullptr, out, 2);
}
